// GraphSAGE_82970178224659
// MI455X (gfx1250) — matrix operand order repaired and checked by reading, not compiled, not run on hardware
//
#include <hip/hip_runtime.h>

typedef __attribute__((ext_vector_type(16))) __bf16 v16bf;
typedef __attribute__((ext_vector_type(8)))  __bf16 v8bf;
typedef __attribute__((ext_vector_type(4)))  __bf16 v4bf;
typedef __attribute__((ext_vector_type(8)))  float  v8f;

#define D_FEAT 128

// Guaranteed-native fp32 global atomic add: the clang::atomic attribute tells
// the backend there is no remote/fine-grained memory involved, so atomicrmw
// fadd lowers to global_atomic_add_f32 instead of a cmpswap retry loop.
__device__ __forceinline__ void atomic_add_f32(float* p, float v) {
#if defined(__has_cpp_attribute) && __has_cpp_attribute(clang::atomic)
    [[clang::atomic(no_remote_memory, no_fine_grained_memory, ignore_denormal_mode)]] {
        __hip_atomic_fetch_add(p, v, __ATOMIC_RELAXED, __HIP_MEMORY_SCOPE_AGENT);
    }
#else
    atomicAdd(p, v);
#endif
}

// ---------------------------------------------------------------- degree
__global__ void deg_count_kernel(const long long* __restrict__ dst,
                                 float* __restrict__ deg, int nEdges) {
    int e = blockIdx.x * blockDim.x + threadIdx.x;
    if (e < nEdges) atomic_add_f32(deg + (int)dst[e], 1.0f);
}

__global__ void deg_inv_kernel(float* __restrict__ deg, int n) {
    int i = blockIdx.x * blockDim.x + threadIdx.x;
    if (i < n) deg[i] = 1.0f / fmaxf(deg[i], 1.0f);
}

// ---------------------------------------------------------------- f32 -> bf16 (x4 per thread)
__global__ void cvt_bf16x4_kernel(const float* __restrict__ in,
                                  __bf16* __restrict__ out, int n4) {
    int i = blockIdx.x * blockDim.x + threadIdx.x;
    if (i < n4) {
        float4 f = ((const float4*)in)[i];
        v4bf o;
        o[0] = (__bf16)f.x; o[1] = (__bf16)f.y;
        o[2] = (__bf16)f.z; o[3] = (__bf16)f.w;
        *(v4bf*)(out + (size_t)i * 4) = o;
    }
}

// ---------------------------------------------------------------- scatter mean (sum part)
// 16 threads per edge; each thread moves 8 bf16 features (one b128 load)
// and does 8 native f32 atomic adds into the fp32 accumulator (L2-resident).
__global__ void scatter_add_kernel(const __bf16* __restrict__ h,
                                   const long long* __restrict__ src,
                                   const long long* __restrict__ dst,
                                   float* __restrict__ agg, int nEdges) {
    int t = blockIdx.x * blockDim.x + threadIdx.x;
    int e = t >> 4;
    if (e >= nEdges) return;
    int c = (t & 15) * 8;
    int s = (int)src[e];
    int d = (int)dst[e];
    v8bf hv = *(const v8bf*)(h + (size_t)s * D_FEAT + c);
    float* ap = agg + (size_t)d * D_FEAT + c;
#pragma unroll
    for (int j = 0; j < 8; ++j) atomic_add_f32(ap + j, (float)hv[j]);
}

// ---------------------------------------------------------------- fused SAGE GEMM
// out[m, n] = sum_k (agg[m,k]*deg_inv[m]) * Wl[n,k] + sum_k h[m,k] * Wr[n,k] + b[n]
// One block = 16-node M tile. 4 waves, each wave owns 2 N-tiles of 16 cols.
// 8 x v_wmma_f32_16x16x32_bf16 per 16x16 output tile (4 K-slices x 2 matrices).
__global__ __launch_bounds__(128)
void sage_gemm_kernel(const float* __restrict__ agg,
                      const __bf16* __restrict__ h,
                      const float* __restrict__ deg_inv,
                      const __bf16* __restrict__ Wl,
                      const __bf16* __restrict__ Wr,
                      const float* __restrict__ bias,
                      __bf16* __restrict__ out_bf,
                      float* __restrict__ out_f,
                      int relu) {
    const int tileM = blockIdx.x * 16;
    const int lane  = threadIdx.x & 31;
    const int wave  = threadIdx.x >> 5;
    const int half  = lane >> 4;     // which half-wave
    const int mrow  = lane & 15;     // row within tile (A) / col within tile (B, C/D)
    const int m     = tileM + mrow;
    const float scale = deg_inv[m];

    // ---- load A operands for all 4 K-slices (per-lane layout:
    //      elems 0..7  -> K = ks*32 + half*8 + j
    //      elems 8..15 -> K = ks*32 + 16 + half*8 + j)
    v16bf aAgg[4], aH[4];
#pragma unroll
    for (int ks = 0; ks < 4; ++ks) {
        const int kb = ks * 32 + half * 8;
        const float* ap = agg + (size_t)m * D_FEAT + kb;
        float4 f0 = *(const float4*)(ap);
        float4 f1 = *(const float4*)(ap + 4);
        float4 f2 = *(const float4*)(ap + 16);
        float4 f3 = *(const float4*)(ap + 20);
        v16bf a;
        a[0]  = (__bf16)(f0.x * scale); a[1]  = (__bf16)(f0.y * scale);
        a[2]  = (__bf16)(f0.z * scale); a[3]  = (__bf16)(f0.w * scale);
        a[4]  = (__bf16)(f1.x * scale); a[5]  = (__bf16)(f1.y * scale);
        a[6]  = (__bf16)(f1.z * scale); a[7]  = (__bf16)(f1.w * scale);
        a[8]  = (__bf16)(f2.x * scale); a[9]  = (__bf16)(f2.y * scale);
        a[10] = (__bf16)(f2.z * scale); a[11] = (__bf16)(f2.w * scale);
        a[12] = (__bf16)(f3.x * scale); a[13] = (__bf16)(f3.y * scale);
        a[14] = (__bf16)(f3.z * scale); a[15] = (__bf16)(f3.w * scale);
        aAgg[ks] = a;

        const __bf16* hp = h + (size_t)m * D_FEAT + kb;
        v8bf h0 = *(const v8bf*)(hp);
        v8bf h1 = *(const v8bf*)(hp + 16);
        v16bf ah;
#pragma unroll
        for (int j = 0; j < 8; ++j) { ah[j] = h0[j]; ah[8 + j] = h1[j]; }
        aH[ks] = ah;
    }

#pragma unroll
    for (int nt = 0; nt < 2; ++nt) {
        const int n = wave * 32 + nt * 16 + mrow;  // column this lane serves
        v8f acc = {};
        // h @ Wr^T : B[k][n] = Wr[n][k], per-lane K = ks*32 + half*8 + [0..7], ks*32 + 16 + half*8 + [0..7]
#pragma unroll
        for (int ks = 0; ks < 4; ++ks) {
            const int kb = ks * 32 + half * 8;
            const __bf16* wp = Wr + (size_t)n * D_FEAT + kb;
            v8bf w0 = *(const v8bf*)(wp);
            v8bf w1 = *(const v8bf*)(wp + 16);
            v16bf bw;
#pragma unroll
            for (int j = 0; j < 8; ++j) { bw[j] = w0[j]; bw[8 + j] = w1[j]; }
            acc = __builtin_amdgcn_wmma_f32_16x16x32_bf16(
                false, aH[ks], false, bw, (short)0, acc, false, false);
        }
        // agg_scaled @ Wl^T
#pragma unroll
        for (int ks = 0; ks < 4; ++ks) {
            const int kb = ks * 32 + half * 8;
            const __bf16* wp = Wl + (size_t)n * D_FEAT + kb;
            v8bf w0 = *(const v8bf*)(wp);
            v8bf w1 = *(const v8bf*)(wp + 16);
            v16bf bw;
#pragma unroll
            for (int j = 0; j < 8; ++j) { bw[j] = w0[j]; bw[8 + j] = w1[j]; }
            acc = __builtin_amdgcn_wmma_f32_16x16x32_bf16(
                false, aAgg[ks], false, bw, (short)0, acc, false, false);
        }

        const float bv = bias[n];
        // C/D layout: elem r -> row = tileM + half*8 + r, col = n
#pragma unroll
        for (int r = 0; r < 8; ++r) {
            float v = acc[r] + bv;
            if (relu) v = fmaxf(v, 0.0f);
            const size_t row = (size_t)(tileM + half * 8 + r);
            if (out_bf) out_bf[row * D_FEAT + n] = (__bf16)v;
            if (out_f)  out_f[row * D_FEAT + n]  = v;
        }
    }
}

// ----------------------------------------------------------------
extern "C" void kernel_launch(void* const* d_in, const int* in_sizes, int n_in,
                              void* d_out, int out_size, void* d_ws, size_t ws_size,
                              hipStream_t stream) {
    const float*     x   = (const float*)d_in[0];
    const long long* ei  = (const long long*)d_in[1];   // [2, E] int64
    // d_in[2] = edge_attr (unused by SAGEConv)
    const float* Wl_in[3] = { (const float*)d_in[3], (const float*)d_in[6], (const float*)d_in[9]  };
    const float* b_in [3] = { (const float*)d_in[4], (const float*)d_in[7], (const float*)d_in[10] };
    const float* Wr_in[3] = { (const float*)d_in[5], (const float*)d_in[8], (const float*)d_in[11] };

    const int NN = in_sizes[0] / D_FEAT;   // 100000
    const int E  = in_sizes[1] / 2;        // 1600000
    const long long* src = ei;
    const long long* dst = ei + E;

    // ---- workspace layout (256B aligned slices)
    char*  ws  = (char*)d_ws;
    size_t off = 0;
    auto carve = [&](size_t bytes) -> void* {
        void* p = ws + off;
        off += (bytes + 255) & ~(size_t)255;
        return p;
    };
    float*  deg  = (float*)carve((size_t)NN * 4);
    __bf16* hA   = (__bf16*)carve((size_t)NN * D_FEAT * 2);
    __bf16* hB   = (__bf16*)carve((size_t)NN * D_FEAT * 2);
    float*  agg  = (float*)carve((size_t)NN * D_FEAT * 4);
    __bf16* Wbf[6];
    for (int i = 0; i < 6; ++i) Wbf[i] = (__bf16*)carve((size_t)D_FEAT * D_FEAT * 2);
    (void)ws_size;

    const int NT = 256;

    // degrees -> deg_inv
    hipMemsetAsync(deg, 0, (size_t)NN * 4, stream);
    deg_count_kernel<<<(E + NT - 1) / NT, NT, 0, stream>>>(dst, deg, E);
    deg_inv_kernel<<<(NN + NT - 1) / NT, NT, 0, stream>>>(deg, NN);

    // bf16 conversions (activations + 6 weight matrices), 4 elems/thread
    {
        const int n4 = NN * D_FEAT / 4;
        cvt_bf16x4_kernel<<<(n4 + NT - 1) / NT, NT, 0, stream>>>(x, hA, n4);
        const int w4 = D_FEAT * D_FEAT / 4;
        for (int L = 0; L < 3; ++L) {
            cvt_bf16x4_kernel<<<(w4 + NT - 1) / NT, NT, 0, stream>>>(Wl_in[L], Wbf[2 * L + 0], w4);
            cvt_bf16x4_kernel<<<(w4 + NT - 1) / NT, NT, 0, stream>>>(Wr_in[L], Wbf[2 * L + 1], w4);
        }
    }

    // three SAGE layers
    __bf16* cur = hA;
    __bf16* nxt = hB;
    for (int L = 0; L < 3; ++L) {
        hipMemsetAsync(agg, 0, (size_t)NN * D_FEAT * 4, stream);
        const long long totScatter = (long long)E * 16;
        scatter_add_kernel<<<(int)((totScatter + NT - 1) / NT), NT, 0, stream>>>(cur, src, dst, agg, E);

        const bool last = (L == 2);
        sage_gemm_kernel<<<NN / 16, 128, 0, stream>>>(
            agg, cur, deg, Wbf[2 * L + 0], Wbf[2 * L + 1], b_in[L],
            last ? nullptr : nxt,
            last ? (float*)d_out : nullptr,
            last ? 0 : 1);

        __bf16* t = cur; cur = nxt; nxt = t;
    }
}
